// Quant_Conv2d_4526895530103
// MI455X (gfx1250) — compile-verified, matrix operand order checked by reading, not run on hardware
//
#include <hip/hip_runtime.h>

// ---------------------------------------------------------------------------
// Quantized 3x3 conv (B=32, Cin=128, Cout=256, H=W=56, pad=1, stride=1)
// via int8 implicit GEMM on v_wmma_i32_16x16x64_iu8 (CDNA5 / gfx1250, wave32)
//
// GEMM view: M = Cout = 256, N = B*H*W = 100,352, K = Cin*9 = 1152 (18 x 64).
// Block (256 thr = 8 wave32): 64 Cout x 128 pixel tile, waves 4(M) x 2(N),
// each wave: 16 x 64 -> four 16x16 i32 accumulators, A fragment reused 4x.
// Double-buffered LDS (A 4KB + B 8KB per buffer), register-staged loads.
// ---------------------------------------------------------------------------

typedef __attribute__((ext_vector_type(8))) int v8i_t;

#define CIN   128
#define COUT  256
#define HDIM  56
#define HW    3136        // 56*56
#define KSZ   1152        // 128*9
#define NPIX  100352      // 32*3136
#define X8_BYTES   12845056   // NPIX*CIN int8
#define WQ_BYTES   294912     // COUT*KSZ int8
#define OUTQ_ELEMS 25690112   // 32*256*56*56

// ---------------------------------------------------------------------------
// Kernel 1: per-output-channel weight quantization + bias quantization.
// Writes wq8 in [cout][tap][cin] layout: a K-chunk (fixed tap, 64 consecutive
// cin) is 64 contiguous bytes.
// ---------------------------------------------------------------------------
__global__ __launch_bounds__(256) void wquant_kernel(
    const float* __restrict__ weight,     // [COUT][CIN][3][3]
    const float* __restrict__ bias,       // [COUT]
    const float* __restrict__ scale_x,    // [1]
    signed char* __restrict__ wq8,        // [COUT][9][CIN]
    int*         __restrict__ bq,         // [COUT]
    float*       __restrict__ scales_out) // [COUT] -> d_out tail
{
  __shared__ float smn[256], smx[256];
  __shared__ float s_sw;
  const int c   = blockIdx.x;
  const int tid = threadIdx.x;
  const float* wrow = weight + (size_t)c * KSZ;

  float mn =  3.4e38f, mx = -3.4e38f;
  for (int i = tid; i < KSZ; i += 256) {
    float v = wrow[i];
    mn = fminf(mn, v);
    mx = fmaxf(mx, v);
  }
  smn[tid] = mn; smx[tid] = mx;
  __syncthreads();
  for (int s = 128; s > 0; s >>= 1) {
    if (tid < s) {
      smn[tid] = fminf(smn[tid], smn[tid + s]);
      smx[tid] = fmaxf(smx[tid], smx[tid + s]);
    }
    __syncthreads();
  }
  if (tid == 0) {
    float sat = fmaxf(fabsf(smn[0]), fabsf(smx[0]));
    float sw  = fmaxf(sat, 1e-8f) / 127.0f;       // symmetric scale, n=127
    float so  = sw * scale_x[0];
    s_sw = sw;
    scales_out[c] = so;
    double v = nearbyint((double)bias[c] / (double)so);  // round-to-even
    v = fmin(fmax(v, -2147483648.0), 2147483647.0);
    bq[c] = (int)(long long)v;
  }
  __syncthreads();
  const float sw = s_sw;
  for (int i = tid; i < KSZ; i += 256) {
    int cin = i / 9;            // source order inside channel: cin-major, 9 taps
    int tap = i - cin * 9;
    float q = rintf(wrow[i] / sw);
    q = fminf(fmaxf(q, -128.0f), 127.0f);
    wq8[(size_t)c * KSZ + (size_t)tap * CIN + cin] = (signed char)(int)q;
  }
}

// ---------------------------------------------------------------------------
// Kernel 2: repack activations int32 NCHW -> int8 NHWC ([pixel][cin]).
// 4x traffic reduction; im2col K-chunks become contiguous 64B runs.
// ---------------------------------------------------------------------------
__global__ __launch_bounds__(256) void pack_kernel(
    const int* __restrict__ xq,          // [B][CIN][H][W] int32 in [-128,127]
    signed char* __restrict__ x8)        // [NPIX][CIN] int8
{
  const int n  = blockIdx.x * 256 + threadIdx.x;   // grid sized exactly
  const int b  = n / HW;
  const int hw = n - b * HW;
  const int* src = xq + (size_t)b * CIN * HW + hw; // stride HW per cin
  int* dst = (int*)(x8 + (size_t)n * CIN);
#pragma unroll 4
  for (int c = 0; c < CIN; c += 4) {
    int p0 = src[(size_t)(c + 0) * HW] & 0xFF;
    int p1 = src[(size_t)(c + 1) * HW] & 0xFF;
    int p2 = src[(size_t)(c + 2) * HW] & 0xFF;
    int p3 = src[(size_t)(c + 3) * HW] & 0xFF;
    dst[c >> 2] = p0 | (p1 << 8) | (p2 << 16) | (p3 << 24);
  }
}

// ---------------------------------------------------------------------------
// Kernel 3: implicit-GEMM int8 conv with WMMA, double-buffered LDS.
// Fragment reads are ds_load_b64 following the 8-bit A-matrix lane swizzle:
//   byte base K(v,hi) = (v&1)*4 + ((v>>1)&1)*16 + (v>>2)*32 + hi*8
// => VGPR pair p covers the aligned 8B run at (p&1)*16 + (p>>1)*32 + hi*8.
// ---------------------------------------------------------------------------
__global__ __launch_bounds__(256) void conv_wmma_kernel(
    const signed char* __restrict__ x8,   // [NPIX][CIN]
    const signed char* __restrict__ wq8,  // [COUT][9][CIN]
    const int* __restrict__ bq,           // [COUT]
    int* __restrict__ out)                // [B][COUT][H][W]
{
  __shared__ __align__(16) signed char Alds[2][64][64];   // [buf][cout][k]
  __shared__ __align__(16) signed char Blds[2][128][64];  // [buf][pixel][k]

  const int pix0  = blockIdx.x * 128;
  const int cout0 = blockIdx.y * 64;
  const int tid   = threadIdx.x;
  const int lane  = tid & 31;
  const int wave  = tid >> 5;
  const int wm    = wave & 3;     // M subtile (16 couts)
  const int wn    = wave >> 2;    // N subtile (64 pixels)
  const int r     = lane & 15;
  const int hi    = lane >> 4;

  // A-loader: thread -> (cout row, 16B segment); 4KB per chunk.
  const int arow = tid >> 2;      // 0..63
  const int aseg = tid & 3;       // 0..3
  // B-loader: thread -> (pixel, 32B half); 8KB per chunk.
  const int brow  = tid >> 1;     // 0..127
  const int bhalf = tid & 1;      // 0..1

  // Decode the B-loader's output pixel once.
  const int bn  = pix0 + brow;
  const int bb  = bn / HW;
  const int bhw = bn - bb * HW;
  const int bh  = bhw / HDIM;
  const int bw  = bhw - bh * HDIM;

  const signed char* aBase = wq8 + (size_t)(cout0 + arow) * KSZ + aseg * 16;

  v8i_t acc[4] = {{}, {}, {}, {}};

  int4 aReg, bReg0, bReg1;

  // --- register-staged global load of one K-chunk ---
  auto loadChunk = [&](int kc) {
    const int tap     = kc >> 1;
    const int kh      = tap / 3;
    const int kw      = tap - kh * 3;
    const int cinBase = (kc & 1) << 6;
    aReg = *(const int4*)(aBase + tap * CIN + cinBase);
    const int hin = bh + kh - 1;
    const int win = bw + kw - 1;
    int4 z = {0, 0, 0, 0};
    bReg0 = z; bReg1 = z;
    if (hin >= 0 && hin < HDIM && win >= 0 && win < HDIM) {
      const signed char* p = x8 + ((size_t)bb * HW + hin * HDIM + win) * CIN +
                             cinBase + bhalf * 32;
      bReg0 = *(const int4*)(p);
      bReg1 = *(const int4*)(p + 16);
    }
  };
  auto storeChunk = [&](int buf) {
    *(int4*)&Alds[buf][arow][aseg * 16]      = aReg;
    *(int4*)&Blds[buf][brow][bhalf * 32]     = bReg0;
    *(int4*)&Blds[buf][brow][bhalf * 32 + 16] = bReg1;
  };

  // Prologue: chunk 0 into buffer 0.
  loadChunk(0);
  storeChunk(0);
  __syncthreads();

  for (int kc = 0; kc < 18; ++kc) {
    const int cur = kc & 1;

    // Issue next chunk's global loads before compute (latency overlap).
    if (kc < 17) {
      loadChunk(kc + 1);
      // Uniform-condition software prefetch of the chunk after next (weights).
      if (kc < 16) {
        const int t2 = (kc + 2) >> 1;
        const int c2 = ((kc + 2) & 1) << 6;
        __builtin_prefetch(aBase + t2 * CIN + c2, 0, 3);
      }
    }

    // --- fragments + 4 WMMAs from current buffer ---
    union { v8i_t v; long long ll[4]; } ua, ub;
    const long long* ap = (const long long*)&Alds[cur][wm * 16 + r][0];
#pragma unroll
    for (int p = 0; p < 4; ++p) {
      const int idx = ((p & 1) << 1) + ((p >> 1) << 2) + hi;  // qword index
      ua.ll[p] = ap[idx];
    }
#pragma unroll
    for (int nf = 0; nf < 4; ++nf) {
      const long long* bp =
          (const long long*)&Blds[cur][wn * 64 + nf * 16 + r][0];
#pragma unroll
      for (int p = 0; p < 4; ++p) {
        const int idx = ((p & 1) << 1) + ((p >> 1) << 2) + hi;
        ub.ll[p] = bp[idx];
      }
      // signed x signed int8, int32 accumulate
      acc[nf] = __builtin_amdgcn_wmma_i32_16x16x64_iu8(true, ua.v, true, ub.v,
                                                       acc[nf], false, false);
    }

    // Stage next chunk into the other buffer.
    if (kc < 17) storeChunk(1 - cur);
    __syncthreads();
  }

  // --- Epilogue: add quantized bias, scatter to NCHW int32 output ---
  // C/D layout: VGPR j, lane l -> M = j + (l>>4)*8, N = l&15.
  const int coutBase = cout0 + wm * 16 + hi * 8;
  int bvals[8];
#pragma unroll
  for (int j = 0; j < 8; ++j) bvals[j] = bq[coutBase + j];
#pragma unroll
  for (int nf = 0; nf < 4; ++nf) {
    const int n  = pix0 + wn * 64 + nf * 16 + r;
    const int b  = n / HW;
    const int hw = n - b * HW;
    int* obase = out + (size_t)b * COUT * HW + (size_t)coutBase * HW + hw;
#pragma unroll
    for (int j = 0; j < 8; ++j) {
      obase[(size_t)j * HW] = acc[nf][j] + bvals[j];
    }
  }
}

// ---------------------------------------------------------------------------
// Launch: d_in = {x_q(int32), scale_x(f32[1]), weight(f32), bias(f32)}
// d_out = out_q (25,690,112 int32) ++ scale_out (256 f32)
// ---------------------------------------------------------------------------
extern "C" void kernel_launch(void* const* d_in, const int* in_sizes, int n_in,
                              void* d_out, int out_size, void* d_ws, size_t ws_size,
                              hipStream_t stream) {
  const int*   x_q     = (const int*)d_in[0];
  const float* scale_x = (const float*)d_in[1];
  const float* weight  = (const float*)d_in[2];
  const float* bias    = (const float*)d_in[3];

  signed char* x8  = (signed char*)d_ws;        // 12,845,056 B
  signed char* wq8 = x8 + X8_BYTES;             //    294,912 B
  int*         bq  = (int*)(wq8 + WQ_BYTES);    //      1,024 B

  int*   out_q  = (int*)d_out;
  float* scales = (float*)((int*)d_out + OUTQ_ELEMS);

  wquant_kernel<<<COUT, 256, 0, stream>>>(weight, bias, scale_x, wq8, bq, scales);
  pack_kernel<<<NPIX / 256, 256, 0, stream>>>(x_q, x8);

  dim3 grid(NPIX / 128, COUT / 64);  // 784 x 4, no tails -> EXEC all-ones
  conv_wmma_kernel<<<grid, 256, 0, stream>>>(x8, wq8, bq, out_q);
}
